// MultiHeadAttention_4011499454945
// MI455X (gfx1250) — matrix operand order repaired and checked by reading, not compiled, not run on hardware
//
#include <hip/hip_runtime.h>
#include <hip/hip_bf16.h>

// ---------------------------------------------------------------------------
// MHA forward for MI455X (gfx1250, wave32, WMMA).
// All matmuls in bf16 WMMA (v_wmma_f32_16x16x32_bf16, f32 accumulate).
// dist (537 MB) dominates HBM traffic; scores strip kept resident in LDS
// (256.5 KB/WGP -- uses CDNA5's 320 KB LDS).
// Mask applied in the softmax sweep (coalesced uchar4), not the WMMA loop.
// 1/sqrt(HD) folded into the q-projection epilogue.
// ---------------------------------------------------------------------------

typedef __attribute__((ext_vector_type(16))) __bf16 bf16x16;
typedef __attribute__((ext_vector_type(8)))  __bf16 bf16x8;
typedef __attribute__((ext_vector_type(8)))  float  f32x8;

#define BATCH   4
#define SEQ     2048
#define DMODEL  512
#define NHEAD   8
#define HEADD   64
#define MROWS   (BATCH * SEQ)          // 8192
#define X_ELEMS (MROWS * DMODEL)       // 4,194,304 floats (x), dist follows

// workspace byte offsets
#define WS_Q    0                       // bf16 [B,H,S,HD]  (pre-scaled by 0.125)
#define WS_K    (8u * 1024 * 1024)      // bf16 [B,H,S,HD]
#define WS_VT   (16u * 1024 * 1024)     // bf16 [B,H,HD,S]
#define WS_ATT  (24u * 1024 * 1024)     // bf16 [B,S,D]

// ---------------------------------------------------------------------------
// GEMM: C[M,512] = (A[M,512] @ W^T + bias) * outScale
// MODE 0: bf16 out [B,H,S,HD]   MODE 1: bf16 out [B,H,HD,S]   MODE 2: f32 [M,512]
// ABF  0: A is f32              ABF 1: A is bf16
// Block = 256 threads = 8 waves (4 row-waves x 2 col-waves), wave tile 32x32.
// Grid = (M/128, 512/64) = (64, 8).
// ---------------------------------------------------------------------------
template <int MODE, int ABF>
__global__ __launch_bounds__(256) void mha_gemm_kernel(
    const float* __restrict__ Af, const __bf16* __restrict__ Ab,
    const float* __restrict__ W,  const float* __restrict__ bias,
    __bf16* __restrict__ outb,    float* __restrict__ outf,
    float outScale)
{
    const int tid   = threadIdx.x;
    const int lane  = tid & 31;
    const int wave  = tid >> 5;
    const int wr    = wave & 3;
    const int wc    = wave >> 2;
    const int l16   = lane & 15;
    const int lh    = lane >> 4;          // 0 or 1 (half-wave)

    const int rowBase = blockIdx.x * 128 + wr * 32;
    const int colBase = blockIdx.y * 64  + wc * 32;

    f32x8 acc[2][2] = {};

    for (int k = 0; k < DMODEL; k += 32) {
        bf16x16 afrag[2];
        bf16x16 bfrag[2];

        // --- A fragments: 16x32 bf16, row = M, e0..7 <-> K=k+lh*8+e, e8..15 <-> +16
        #pragma unroll
        for (int r = 0; r < 2; ++r) {
            const int row = rowBase + r * 16 + l16;
            const int k0  = k + lh * 8;
            if (ABF) {
                bf16x8 lo = *(const bf16x8*)(Ab + row * DMODEL + k0);
                bf16x8 hi = *(const bf16x8*)(Ab + row * DMODEL + k0 + 16);
                #pragma unroll
                for (int i = 0; i < 8; ++i) { afrag[r][i] = lo[i]; afrag[r][8 + i] = hi[i]; }
            } else {
                const float4 p0 = *(const float4*)(Af + row * DMODEL + k0);
                const float4 p1 = *(const float4*)(Af + row * DMODEL + k0 + 4);
                const float4 p2 = *(const float4*)(Af + row * DMODEL + k0 + 16);
                const float4 p3 = *(const float4*)(Af + row * DMODEL + k0 + 20);
                afrag[r][0]  = (__bf16)p0.x; afrag[r][1]  = (__bf16)p0.y;
                afrag[r][2]  = (__bf16)p0.z; afrag[r][3]  = (__bf16)p0.w;
                afrag[r][4]  = (__bf16)p1.x; afrag[r][5]  = (__bf16)p1.y;
                afrag[r][6]  = (__bf16)p1.z; afrag[r][7]  = (__bf16)p1.w;
                afrag[r][8]  = (__bf16)p2.x; afrag[r][9]  = (__bf16)p2.y;
                afrag[r][10] = (__bf16)p2.z; afrag[r][11] = (__bf16)p2.w;
                afrag[r][12] = (__bf16)p3.x; afrag[r][13] = (__bf16)p3.y;
                afrag[r][14] = (__bf16)p3.z; afrag[r][15] = (__bf16)p3.w;
            }
        }

        // --- B fragments: 32x16 bf16, B[kk][n] = W[n][k+kk] (C = A @ W^T)
        //     lane col n = l16, e0..7 <-> kk = lh*8 + e, e8..15 <-> +16  -> two 8-float runs of W row n
        #pragma unroll
        for (int c = 0; c < 2; ++c) {
            const int n  = colBase + c * 16 + l16;
            const int kk = k + lh * 8;
            const float* wp = W + n * DMODEL + kk;
            __builtin_prefetch((const void*)(wp + 32), 0, 1);   // global_prefetch_b8
            const float4 q0 = *(const float4*)(wp);
            const float4 q1 = *(const float4*)(wp + 4);
            const float4 q2 = *(const float4*)(wp + 16);
            const float4 q3 = *(const float4*)(wp + 20);
            bfrag[c][0]  = (__bf16)q0.x; bfrag[c][1]  = (__bf16)q0.y;
            bfrag[c][2]  = (__bf16)q0.z; bfrag[c][3]  = (__bf16)q0.w;
            bfrag[c][4]  = (__bf16)q1.x; bfrag[c][5]  = (__bf16)q1.y;
            bfrag[c][6]  = (__bf16)q1.z; bfrag[c][7]  = (__bf16)q1.w;
            bfrag[c][8]  = (__bf16)q2.x; bfrag[c][9]  = (__bf16)q2.y;
            bfrag[c][10] = (__bf16)q2.z; bfrag[c][11] = (__bf16)q2.w;
            bfrag[c][12] = (__bf16)q3.x; bfrag[c][13] = (__bf16)q3.y;
            bfrag[c][14] = (__bf16)q3.z; bfrag[c][15] = (__bf16)q3.w;
        }

        #pragma unroll
        for (int r = 0; r < 2; ++r)
            #pragma unroll
            for (int c = 0; c < 2; ++c)
                acc[r][c] = __builtin_amdgcn_wmma_f32_16x16x32_bf16(
                    false, afrag[r], false, bfrag[c], (short)0, acc[r][c], false, false);
    }

    // --- epilogue: C/D layout -> element i: row = base + lh*8 + i, col = base + l16
    #pragma unroll
    for (int r = 0; r < 2; ++r) {
        #pragma unroll
        for (int c = 0; c < 2; ++c) {
            const int col = colBase + c * 16 + l16;
            const float bv = bias[col];
            #pragma unroll
            for (int i = 0; i < 8; ++i) {
                const int row = rowBase + r * 16 + lh * 8 + i;
                const float v = (acc[r][c][i] + bv) * outScale;
                if (MODE == 2) {
                    outf[row * DMODEL + col] = v;
                } else {
                    const int b = row >> 11, s = row & (SEQ - 1);
                    const int h = col >> 6,  d = col & (HEADD - 1);
                    if (MODE == 0)
                        outb[(((b * NHEAD + h) * SEQ) + s) * HEADD + d] = (__bf16)v;
                    else
                        outb[(((b * NHEAD + h) * HEADD) + d) * SEQ + s] = (__bf16)v;
                }
            }
        }
    }
}

// ---------------------------------------------------------------------------
// Attention: one WG = one (b,h) + 32 query rows; 8 waves / 256 threads.
// LDS: 32 x (2048+4) f32 score strip (256.5 KB) + 32 f32 row-reciprocals.
// ---------------------------------------------------------------------------
#define LDW        (SEQ + 4)
#define SMEM_F32   (32 * LDW + 32)
#define SMEM_BYTES (SMEM_F32 * 4)

__global__ __launch_bounds__(256) void mha_attn_kernel(
    const __bf16* __restrict__ qb, const __bf16* __restrict__ kb,
    const __bf16* __restrict__ vt, const unsigned char* __restrict__ mask,
    float* __restrict__ dist, __bf16* __restrict__ attnout)
{
    extern __shared__ float smem[];
    float* sc   = smem;                 // [32][LDW]
    float* rinv = smem + 32 * LDW;      // [32]

    const int tid  = threadIdx.x;
    const int lane = tid & 31;
    const int wave = tid >> 5;
    const int l16  = lane & 15;
    const int lh   = lane >> 4;

    const int nQB   = SEQ / 32;
    const int bh    = blockIdx.x / nQB;           // b*NHEAD + h
    const int qrow0 = (blockIdx.x % nQB) * 32;
    const int b     = bh / NHEAD;
    const int h     = bh % NHEAD;

    // ---- Q fragments: held in registers for the whole j sweep -------------
    bf16x16 afr[2][2];   // [row-tile][k-chunk of HD]
    #pragma unroll
    for (int rt = 0; rt < 2; ++rt) {
        #pragma unroll
        for (int kc = 0; kc < 2; ++kc) {
            const int row = qrow0 + rt * 16 + l16;
            const int k0  = kc * 32 + lh * 16;
            const __bf16* qp = qb + ((long)bh * SEQ + row) * HEADD + k0;
            bf16x8 lo = *(const bf16x8*)(qp);
            bf16x8 hi = *(const bf16x8*)(qp + 8);
            #pragma unroll
            for (int i = 0; i < 8; ++i) { afr[rt][kc][i] = lo[i]; afr[rt][kc][8 + i] = hi[i]; }
        }
    }

    // ---- Phase 1: raw scores = q k^T into LDS (q pre-scaled by 1/sqrt(HD))
    // 2 row-tiles x 128 col-tiles = 256 tiles of 16x16, round-robined over 8 waves.
    for (int t = wave; t < 256; t += 8) {
        const int rt = t & 1;
        const int ct = t >> 1;
        const int j0 = ct * 16;

        f32x8 acc = {};
        #pragma unroll
        for (int kc = 0; kc < 2; ++kc) {
            const int n  = j0 + l16;
            const int kk = kc * 32 + lh * 16;
            bf16x16 bfr = *(const bf16x16*)(kb + ((long)bh * SEQ + n) * HEADD + kk);
            acc = __builtin_amdgcn_wmma_f32_16x16x32_bf16(
                false, afr[rt][kc], false, bfr, (short)0, acc, false, false);
        }

        const int col = j0 + l16;
        #pragma unroll
        for (int i = 0; i < 8; ++i) {
            const int rloc = rt * 16 + lh * 8 + i;
            sc[rloc * LDW + col] = acc[i];
        }
    }
    __syncthreads();

    // ---- Phase 2: mask (coalesced uchar4) + max, then exp + sum -----------
    // 8 threads per row, shuffle-reduce in groups of 8.
    {
        const int row = tid >> 3;
        const int sub = tid & 7;
        float* rp = sc + row * LDW;
        const uchar4* mp = (const uchar4*)(mask + ((long)b * SEQ + qrow0 + row) * SEQ);

        float mx = -3.0e38f;
        for (int c = sub * 4; c < SEQ; c += 32) {
            float4 v = *(const float4*)(rp + c);
            const uchar4 m = mp[c >> 2];
            v.x = m.x ? -1e10f : v.x;
            v.y = m.y ? -1e10f : v.y;
            v.z = m.z ? -1e10f : v.z;
            v.w = m.w ? -1e10f : v.w;
            *(float4*)(rp + c) = v;
            mx = fmaxf(mx, fmaxf(fmaxf(v.x, v.y), fmaxf(v.z, v.w)));
        }
        mx = fmaxf(mx, __shfl_xor(mx, 1, 8));
        mx = fmaxf(mx, __shfl_xor(mx, 2, 8));
        mx = fmaxf(mx, __shfl_xor(mx, 4, 8));

        float sum = 0.0f;
        for (int c = sub * 4; c < SEQ; c += 32) {
            float4 v = *(const float4*)(rp + c);
            v.x = __expf(v.x - mx); v.y = __expf(v.y - mx);
            v.z = __expf(v.z - mx); v.w = __expf(v.w - mx);
            sum += v.x + v.y + v.z + v.w;
            *(float4*)(rp + c) = v;
        }
        sum += __shfl_xor(sum, 1, 8);
        sum += __shfl_xor(sum, 2, 8);
        sum += __shfl_xor(sum, 4, 8);
        if (sub == 0) rinv[row] = 1.0f / sum;
    }
    __syncthreads();

    // ---- Phase 3a: write normalized dist (537 MB stream -> HBM) -----------
    {
        const int row = tid >> 3;
        const int sub = tid & 7;
        const float ri = rinv[row];
        float* rp = sc + row * LDW;
        float* dp = dist + (((long)bh * SEQ) + qrow0 + row) * SEQ;
        for (int c = sub * 4; c < SEQ; c += 32) {
            float4 v = *(const float4*)(rp + c);
            v.x *= ri; v.y *= ri; v.z *= ri; v.w *= ri;
            *(float4*)(dp + c) = v;
        }
    }

    // ---- Phase 3b: attn = P @ V  (K-dim = 2048, 64 WMMA steps per wave) ---
    // 2 row-tiles x 4 d-tiles = 8 tiles -> one per wave.
    {
        const int rt = wave >> 2;
        const int cd = wave & 3;
        f32x8 acc = {};

        for (int jc = 0; jc < SEQ; jc += 32) {
            // A: P tile from LDS (f32 -> bf16)
            bf16x16 af;
            {
                const int row = rt * 16 + l16;
                const float* pp = sc + row * LDW + jc + lh * 16;
                const float4 p0 = *(const float4*)(pp);
                const float4 p1 = *(const float4*)(pp + 4);
                const float4 p2 = *(const float4*)(pp + 8);
                const float4 p3 = *(const float4*)(pp + 12);
                af[0]  = (__bf16)p0.x; af[1]  = (__bf16)p0.y;
                af[2]  = (__bf16)p0.z; af[3]  = (__bf16)p0.w;
                af[4]  = (__bf16)p1.x; af[5]  = (__bf16)p1.y;
                af[6]  = (__bf16)p1.z; af[7]  = (__bf16)p1.w;
                af[8]  = (__bf16)p2.x; af[9]  = (__bf16)p2.y;
                af[10] = (__bf16)p2.z; af[11] = (__bf16)p2.w;
                af[12] = (__bf16)p3.x; af[13] = (__bf16)p3.y;
                af[14] = (__bf16)p3.z; af[15] = (__bf16)p3.w;
            }
            // B: V^T [B,H,HD,S] -> contiguous 32B bf16 loads
            const int d  = cd * 16 + l16;
            const int kk = jc + lh * 16;
            bf16x16 bf = *(const bf16x16*)(vt + ((long)bh * HEADD + d) * SEQ + kk);

            acc = __builtin_amdgcn_wmma_f32_16x16x32_bf16(
                false, af, false, bf, (short)0, acc, false, false);
        }

        const int dcol = cd * 16 + l16;
        #pragma unroll
        for (int i = 0; i < 8; ++i) {
            const int rloc = rt * 16 + lh * 8 + i;
            const float v = acc[i] * rinv[rloc];
            const int srow = qrow0 + rloc;
            attnout[((long)b * SEQ + srow) * DMODEL + h * HEADD + dcol] = (__bf16)v;
        }
    }
}

// ---------------------------------------------------------------------------
extern "C" void kernel_launch(void* const* d_in, const int* in_sizes, int n_in,
                              void* d_out, int out_size, void* d_ws, size_t ws_size,
                              hipStream_t stream)
{
    const float* Q   = (const float*)d_in[0];
    const float* K   = (const float*)d_in[1];
    const float* V   = (const float*)d_in[2];
    const unsigned char* mask = (const unsigned char*)d_in[3];
    const float* Wq = (const float*)d_in[4];
    const float* bq = (const float*)d_in[5];
    const float* Wk = (const float*)d_in[6];
    const float* bk = (const float*)d_in[7];
    const float* Wv = (const float*)d_in[8];
    const float* bv = (const float*)d_in[9];
    const float* Wo = (const float*)d_in[10];
    const float* bo = (const float*)d_in[11];

    float* x    = (float*)d_out;
    float* dist = (float*)d_out + X_ELEMS;

    char* ws = (char*)d_ws;
    __bf16* qbuf  = (__bf16*)(ws + WS_Q);
    __bf16* kbuf  = (__bf16*)(ws + WS_K);
    __bf16* vtbuf = (__bf16*)(ws + WS_VT);
    __bf16* abuf  = (__bf16*)(ws + WS_ATT);

    const dim3 gGemm(MROWS / 128, DMODEL / 64);
    const dim3 bGemm(256);

    // Q/K/V projections (bf16 WMMA, f32 accumulate); q pre-scaled by 1/sqrt(HD)
    mha_gemm_kernel<0, 0><<<gGemm, bGemm, 0, stream>>>(Q, nullptr, Wq, bq, qbuf,  nullptr, 0.125f);
    mha_gemm_kernel<0, 0><<<gGemm, bGemm, 0, stream>>>(K, nullptr, Wk, bk, kbuf,  nullptr, 1.0f);
    mha_gemm_kernel<1, 0><<<gGemm, bGemm, 0, stream>>>(V, nullptr, Wv, bv, vtbuf, nullptr, 1.0f);

    // attention core (needs 256.5 KB dynamic LDS -- CDNA5 WGP has 320 KB)
    hipFuncSetAttribute((const void*)mha_attn_kernel,
                        hipFuncAttributeMaxDynamicSharedMemorySize, SMEM_BYTES);
    const dim3 gAttn(BATCH * NHEAD * (SEQ / 32));
    mha_attn_kernel<<<gAttn, dim3(256), SMEM_BYTES, stream>>>(qbuf, kbuf, vtbuf, mask, dist, abuf);

    // output projection -> f32 x
    mha_gemm_kernel<2, 1><<<gGemm, bGemm, 0, stream>>>(nullptr, abuf, Wo, bo, nullptr, x, 1.0f);
}
